// ReconstructionDecoder_37168646980129
// MI455X (gfx1250) — matrix operand order repaired and checked by reading, not compiled, not run on hardware
//
#include <hip/hip_runtime.h>
#include <hip/hip_bf16.h>
#include <stdint.h>

typedef __bf16 bf16;
typedef __attribute__((ext_vector_type(16))) bf16  v16bf;
typedef __attribute__((ext_vector_type(8)))  float v8f;

constexpr int kB = 128, kS = 512, kH = 512, kO = 64;
constexpr int kKin = 1536;   // concat: ctx | xt | h
constexpr int kG   = 4 * kH; // 2048 gate outputs

union BfFrag { v16bf v; uint4 q[2]; unsigned short u[16]; };

__device__ __forceinline__ unsigned short f2bf(float f) {
  union { float f; unsigned u; } x; x.f = f;
  unsigned u = x.u;
  unsigned r = (u + 0x7fffu + ((u >> 16) & 1u)) >> 16;  // RNE
  return (unsigned short)r;
}
__device__ __forceinline__ float bf2f(unsigned short b) {
  union { unsigned u; float f; } x; x.u = ((unsigned)b) << 16;
  return x.f;
}
__device__ __forceinline__ float sigmoidf_(float x) { return 1.f / (1.f + __expf(-x)); }

__device__ __forceinline__ v8f wmma_bf16(v16bf a, v16bf b, v8f c) {
  return __builtin_amdgcn_wmma_f32_16x16x32_bf16(false, a, false, b, (short)0, c, false, false);
}

// A fragment (16x32 bf16) from row-major [M x K] global matrix
__device__ __forceinline__ v16bf load_a(const unsigned short* A, int lda, int r0, int k0, int lane) {
  int r  = r0 + (lane & 15);
  int kb = k0 + ((lane >> 4) << 3);
  const unsigned short* p = A + (long)r * lda + kb;
  BfFrag f;
  f.q[0] = *(const uint4*)(p);        // K = kb .. kb+7
  f.q[1] = *(const uint4*)(p + 16);   // K = kb+16 .. kb+23
  return f.v;
}
// B fragment (32x16 bf16) where B^T is the row-major weight W [N x K]
__device__ __forceinline__ v16bf load_b(const unsigned short* W, int ldw, int n0, int k0, int lane) {
  int n  = n0 + (lane & 15);
  int kb = k0 + ((lane >> 4) << 3);
  const unsigned short* p = W + (long)n * ldw + kb;
  BfFrag f;
  f.q[0] = *(const uint4*)(p);        // K = kb .. kb+7
  f.q[1] = *(const uint4*)(p + 16);   // K = kb+16 .. kb+23
  return f.v;
}

// ---------------- setup kernels ----------------
__global__ void cvt_f32_bf16(const float* __restrict__ src, unsigned short* __restrict__ dst,
                             int rows, int cols, int sld, int soff, int dld, int doff) {
  long i = (long)blockIdx.x * blockDim.x + threadIdx.x;
  long n = (long)rows * cols;
  if (i >= n) return;
  int r = (int)(i / cols), c = (int)(i % cols);
  dst[(long)r * dld + doff + c] = f2bf(src[(long)r * sld + soff + c]);
}

__global__ void copy_f32(const float* __restrict__ src, float* __restrict__ dst, int n) {
  int i = blockIdx.x * blockDim.x + threadIdx.x;
  if (i < n) dst[i] = src[i];
}

// generic bf16 GEMM, bf16 output (used for enc_proj precompute)
__global__ void __launch_bounds__(256)
gemm_bf16_bf16out(const unsigned short* __restrict__ A, int lda,
                  const unsigned short* __restrict__ W, int ldw,
                  unsigned short* __restrict__ C, int ldc,
                  int Mtiles, int Ntiles, int K) {
  int lane = threadIdx.x & 31;
  int tid  = (int)((blockIdx.x * (long)blockDim.x + threadIdx.x) >> 5);
  if (tid >= Mtiles * Ntiles) return;
  int nt = tid % Ntiles, mt = tid / Ntiles;
  v8f acc;
#pragma unroll
  for (int v = 0; v < 8; ++v) acc[v] = 0.f;
  for (int k0 = 0; k0 < K; k0 += 32) {
    v16bf a = load_a(A, lda, mt * 16, k0, lane);
    v16bf b = load_b(W, ldw, nt * 16, k0, lane);
    acc = wmma_bf16(a, b, acc);
  }
  int n  = nt * 16 + (lane & 15);
  int mb = mt * 16 + ((lane >> 4) << 3);
#pragma unroll
  for (int v = 0; v < 8; ++v) C[(long)(mb + v) * ldc + n] = f2bf(acc[v]);
}

// ---------------- per-step kernel 1: attention ----------------
// One workgroup per batch. Phase1: e_s = Wa2 . tanh(enc_proj[b,s,:] + hid_proj[b,:])
// Phase2: softmax over s (in LDS), ctx[b,:] = sum_s attn_s * enc[b,s,:] -> bf16 into inp ctx slot.
__global__ void __launch_bounds__(256)
attn_kernel(const unsigned short* __restrict__ encproj,
            const unsigned short* __restrict__ enc,
            const float* __restrict__ hidproj,
            const float* __restrict__ Wa2,
            unsigned short* __restrict__ inp) {
  __shared__ float sHid[kH];
  __shared__ float sW[kH];
  __shared__ float sE[kS];
  __shared__ float sR[256];
  int tid = threadIdx.x;
  int b   = blockIdx.x;
  sHid[tid]       = hidproj[b * kH + tid];
  sHid[tid + 256] = hidproj[b * kH + tid + 256];
  sW[tid]         = Wa2[tid];
  sW[tid + 256]   = Wa2[tid + 256];
  __syncthreads();
  int wave = tid >> 5, lane = tid & 31;
  for (int s = wave; s < kS; s += 8) {
    const unsigned short* row = encproj + (long)(b * kS + s) * kH;
    float p = 0.f;
#pragma unroll
    for (int kk = 0; kk < 8; ++kk) {
      int h0 = kk * 64 + lane * 2;
      unsigned pv = *(const unsigned*)(row + h0);
      float e0 = tanhf(bf2f((unsigned short)pv) + sHid[h0]);
      float e1 = tanhf(bf2f((unsigned short)(pv >> 16)) + sHid[h0 + 1]);
      p += sW[h0] * e0 + sW[h0 + 1] * e1;
    }
#pragma unroll
    for (int off = 16; off; off >>= 1) p += __shfl_xor(p, off, 32);
    if (lane == 0) sE[s] = p;
  }
  __syncthreads();
  // block max
  sR[tid] = fmaxf(sE[tid], sE[tid + 256]);
  __syncthreads();
  for (int off = 128; off; off >>= 1) {
    if (tid < off) sR[tid] = fmaxf(sR[tid], sR[tid + off]);
    __syncthreads();
  }
  float mx = sR[0];
  __syncthreads();
  float p0 = __expf(sE[tid] - mx), p1 = __expf(sE[tid + 256] - mx);
  sE[tid] = p0; sE[tid + 256] = p1;
  sR[tid] = p0 + p1;
  __syncthreads();
  for (int off = 128; off; off >>= 1) {
    if (tid < off) sR[tid] += sR[tid + off];
    __syncthreads();
  }
  float inv = 1.f / sR[0];
  // phase 2: ctx
  float a0 = 0.f, a1 = 0.f;
  const unsigned short* ep = enc + (long)b * kS * kH + tid * 2;
  for (int s = 0; s < kS; ++s) {
    float w = sE[s] * inv;
    unsigned pv = *(const unsigned*)(ep + (long)s * kH);
    a0 += w * bf2f((unsigned short)pv);
    a1 += w * bf2f((unsigned short)(pv >> 16));
  }
  inp[b * kKin + tid * 2]     = f2bf(a0);
  inp[b * kKin + tid * 2 + 1] = f2bf(a1);
}

// ---------------- per-step kernel 2: fused LSTM cell ----------------
// 8 blocks of 512 threads (16 waves); block handles 16 batches.
// gates = [ctx|xt|h] @ Wcat^T (WMMA, K=1536) -> pointwise LSTM on fragments ->
// h2 to LDS + inp(xt,h) -> y + next hid_proj GEMM (WMMA, K=512) from LDS.
// t < 0: init path, compute hid_proj from current h only.
__global__ void __launch_bounds__(512)
lstm_step(unsigned short* __restrict__ inp,        // [kB][1536] ctx|xt|h (bf16)
          const unsigned short* __restrict__ Wcat, // [2048][1536]
          const float* __restrict__ bih, const float* __restrict__ bhh,
          float* __restrict__ cbuf,                // [kB][kH]
          const unsigned short* __restrict__ Wyh,  // [576][512] rows 0..63 Wout, 64.. Wa1_hid
          const float* __restrict__ bout,          // [64]
          float* __restrict__ out,                 // [kB][kS][kO]
          float* __restrict__ hidproj,             // [kB][kH]
          int t) {
  __shared__ unsigned short sA[16 * kKin];  // 48 KB
  __shared__ unsigned short sH[16 * kH];    // 16 KB
  int tid  = threadIdx.x;
  int lane = tid & 31, wave = tid >> 5;
  int b0   = blockIdx.x * 16;
  int mb   = (lane >> 4) << 3;

  if (t >= 0) {
    for (int i = tid; i < 16 * kKin / 8; i += 512) {
      int r = i / (kKin / 8), c = i % (kKin / 8);
      *(uint4*)&sA[r * kKin + c * 8] = *(const uint4*)&inp[(long)(b0 + r) * kKin + c * 8];
    }
    __syncthreads();
    for (int jj = 0; jj < 2; ++jj) {
      int nI = wave * 16 + jj * 256;   // i-gate column base; f/g/o at +512/+1024/+1536
      v8f aI, aF, aG, aO;
#pragma unroll
      for (int v = 0; v < 8; ++v) { aI[v] = 0.f; aF[v] = 0.f; aG[v] = 0.f; aO[v] = 0.f; }
      for (int k0 = 0; k0 < kKin; k0 += 32) {
        int r  = lane & 15;
        int kb = k0 + ((lane >> 4) << 3);
        BfFrag fa;
        fa.q[0] = *(const uint4*)&sA[r * kKin + kb];
        fa.q[1] = *(const uint4*)&sA[r * kKin + kb + 16];
        v16bf bI = load_b(Wcat, kKin, nI,        k0, lane);
        v16bf bF = load_b(Wcat, kKin, nI +  512, k0, lane);
        v16bf bG = load_b(Wcat, kKin, nI + 1024, k0, lane);
        v16bf bO = load_b(Wcat, kKin, nI + 1536, k0, lane);
        aI = wmma_bf16(fa.v, bI, aI);
        aF = wmma_bf16(fa.v, bF, aF);
        aG = wmma_bf16(fa.v, bG, aG);
        aO = wmma_bf16(fa.v, bO, aO);
      }
      int col = nI + (lane & 15);
      float bI_ = bih[col]        + bhh[col];
      float bF_ = bih[col +  512] + bhh[col +  512];
      float bG_ = bih[col + 1024] + bhh[col + 1024];
      float bO_ = bih[col + 1536] + bhh[col + 1536];
#pragma unroll
      for (int v = 0; v < 8; ++v) {
        int bidx = b0 + mb + v;
        float iv = sigmoidf_(aI[v] + bI_);
        float fv = sigmoidf_(aF[v] + bF_);
        float gv = tanhf(aG[v] + bG_);
        float ov = sigmoidf_(aO[v] + bO_);
        float cold = cbuf[(long)bidx * kH + col];
        float cn = fv * cold + iv * gv;
        cbuf[(long)bidx * kH + col] = cn;
        float h2 = ov * tanhf(cn);
        unsigned short hb = f2bf(h2);
        inp[(long)bidx * kKin +  512 + col] = hb;  // xt for next step
        inp[(long)bidx * kKin + 1024 + col] = hb;  // h  for next step
        sH[(mb + v) * kH + col] = hb;
      }
    }
    __syncthreads();
  } else {
    for (int i = tid; i < 16 * kH / 8; i += 512) {
      int r = i / (kH / 8), c = i % (kH / 8);
      *(uint4*)&sH[r * kH + c * 8] = *(const uint4*)&inp[(long)(b0 + r) * kKin + 1024 + c * 8];
    }
    __syncthreads();
  }

  // y (tiles 0..3) + hid_proj (tiles 4..35): N=576, K=512, A = sH (h2, bf16)
  for (int nt = wave; nt < 36; nt += 16) {
    if (t < 0 && nt < 4) continue;
    v8f acc;
#pragma unroll
    for (int v = 0; v < 8; ++v) acc[v] = 0.f;
    for (int k0 = 0; k0 < kH; k0 += 32) {
      int r  = lane & 15;
      int kb = k0 + ((lane >> 4) << 3);
      BfFrag fa;
      fa.q[0] = *(const uint4*)&sH[r * kH + kb];
      fa.q[1] = *(const uint4*)&sH[r * kH + kb + 16];
      v16bf bb = load_b(Wyh, kH, nt * 16, k0, lane);
      acc = wmma_bf16(fa.v, bb, acc);
    }
    int cc = lane & 15;
    if (nt < 4) {
      int colo = nt * 16 + cc;
      float bo = bout[colo];
#pragma unroll
      for (int v = 0; v < 8; ++v)
        out[(long)(b0 + mb + v) * kS * kO + (long)(kS - 1 - t) * kO + colo] = acc[v] + bo;
    } else {
      int colh = (nt - 4) * 16 + cc;
#pragma unroll
      for (int v = 0; v < 8; ++v)
        hidproj[(long)(b0 + mb + v) * kH + colh] = acc[v];
    }
  }
}

extern "C" void kernel_launch(void* const* d_in, const int* in_sizes, int n_in,
                              void* d_out, int out_size, void* d_ws, size_t ws_size,
                              hipStream_t stream) {
  const float* x    = (const float*)d_in[0];
  const float* h0   = (const float*)d_in[1];
  const float* c0   = (const float*)d_in[2];
  const float* enc  = (const float*)d_in[3];
  const float* Wa1  = (const float*)d_in[5];
  const float* Wa2  = (const float*)d_in[6];
  const float* Wih  = (const float*)d_in[7];
  const float* Whh  = (const float*)d_in[8];
  const float* bih  = (const float*)d_in[9];
  const float* bhh  = (const float*)d_in[10];
  const float* Wout = (const float*)d_in[11];
  const float* bout = (const float*)d_in[12];
  float* out = (float*)d_out;

  char* p = (char*)d_ws;
  auto alloc = [&](size_t bytes) { char* r = p; p += (bytes + 255) & ~(size_t)255; return r; };
  unsigned short* enc_bf   = (unsigned short*)alloc((size_t)kB * kS * kH * 2);
  unsigned short* eproj_bf = (unsigned short*)alloc((size_t)kB * kS * kH * 2);
  unsigned short* Wenc_bf  = (unsigned short*)alloc((size_t)kH * kH * 2);
  unsigned short* Wcat     = (unsigned short*)alloc((size_t)kG * kKin * 2);
  unsigned short* Wyh      = (unsigned short*)alloc((size_t)(kO + kH) * kH * 2);
  unsigned short* inp      = (unsigned short*)alloc((size_t)kB * kKin * 2);
  float* cbuf    = (float*)alloc((size_t)kB * kH * 4);
  float* hidproj = (float*)alloc((size_t)kB * kH * 4);

  auto cvt = [&](const float* s, unsigned short* d, int rows, int cols,
                 int sld, int soff, int dld, int doff) {
    long n = (long)rows * cols;
    int blocks = (int)((n + 255) / 256);
    cvt_f32_bf16<<<blocks, 256, 0, stream>>>(s, d, rows, cols, sld, soff, dld, doff);
  };
  // bf16 packing (re-done every call; negligible vs the 512-step loop)
  cvt(enc,  enc_bf,  kB * kS, kH, kH, 0, kH, 0);
  cvt(Wa1,  Wenc_bf, kH, kH, 2 * kH, 0, kH, 0);                 // Wa1[:, :H]
  cvt(Wout, Wyh, kO, kH, kH, 0, kH, 0);                         // Wyh rows 0..63
  cvt(Wa1,  Wyh + (size_t)kO * kH, kH, kH, 2 * kH, kH, kH, 0);  // Wa1[:, H:] rows 64..575
  cvt(Wih,  Wcat, kG, 2 * kH, 2 * kH, 0, kKin, 0);              // cols 0..1023
  cvt(Whh,  Wcat, kG, kH, kH, 0, kKin, 2 * kH);                 // cols 1024..1535
  cvt(x,    inp, kB, kH, kH, 0, kKin, kH);                      // xt slot
  cvt(h0,   inp, kB, kH, kH, 0, kKin, 2 * kH);                  // h slot
  copy_f32<<<(kB * kH + 255) / 256, 256, 0, stream>>>(c0, cbuf, kB * kH);

  // enc_proj = enc @ Wa1[:, :H].T   (M=65536, N=512, K=512), stored bf16
  {
    int Mtiles = (kB * kS) / 16, Ntiles = kH / 16;
    long tiles = (long)Mtiles * Ntiles;
    int blocks = (int)((tiles + 7) / 8);
    gemm_bf16_bf16out<<<blocks, 256, 0, stream>>>(enc_bf, kH, Wenc_bf, kH,
                                                  eproj_bf, kH, Mtiles, Ntiles, kH);
  }
  // initial hid_proj from h0
  lstm_step<<<8, 512, 0, stream>>>(inp, Wcat, bih, bhh, cbuf, Wyh, bout, out, hidproj, -1);
  // sequential decode
  for (int t = 0; t < kS; ++t) {
    attn_kernel<<<kB, 256, 0, stream>>>(eproj_bf, enc_bf, hidproj, Wa2, inp);
    lstm_step<<<8, 512, 0, stream>>>(inp, Wcat, bih, bhh, cbuf, Wyh, bout, out, hidproj, t);
  }
}
